// MMT_VideoQA_45518063403410
// MI455X (gfx1250) — matrix operand order repaired and checked by reading, not compiled, not run on hardware
//
#include <hip/hip_runtime.h>
#include <hip/hip_bf16.h>
#include <math.h>

typedef _Float16 half_t;
typedef __attribute__((ext_vector_type(16))) _Float16 v16h;
typedef __attribute__((ext_vector_type(8)))  _Float16 v8h;
typedef __attribute__((ext_vector_type(8)))  float    v8f;
#if defined(__gfx1250__)
typedef __attribute__((address_space(3))) half_t lds_half_t;
#endif

#define B_   64
#define Q_   64
#define T_   448
#define D_   512
#define FD_  1024
#define H_   8
#define NL_  2
#define FF_  2048
#define S_   512
#define DH_  64
#define BS_  (B_*S_)   /* 32768 */
#define BT_  (B_*T_)   /* 28672 */
#define EPS_ 1e-12f

// ---------------------------------------------------------------------------
// CDNA5 async global->LDS copy (ASYNCcnt path), 16B per lane.
// ---------------------------------------------------------------------------
__device__ __forceinline__ void async_copy_b128(half_t* lds_dst, const half_t* gsrc) {
#if defined(__gfx1250__)
  asm volatile("global_load_async_to_lds_b128 %0, %1, off"
               :: "v"((lds_half_t*)lds_dst), "v"(gsrc)
               : "memory");
#else
  *(v8h*)lds_dst = *(const v8h*)gsrc;
#endif
}
__device__ __forceinline__ void async_wait0() {
#if defined(__gfx1250__)
  asm volatile("s_wait_asynccnt 0x0" ::: "memory");
#endif
}

// ---------------------------------------------------------------------------
// WMMA fragment loaders (CDNA5 16-bit layouts, wave32)
// A 16x32: lane holds row m=lane%16; halves = two 8-half chunks at
//          k + 8*half and k + 8*half + 16.
// B 32x16: lane holds col n=lane%16; two 8-half chunks at k + 8*half and k + 8*half + 16.
// ---------------------------------------------------------------------------
__device__ __forceinline__ v16h ldfragA(const half_t* p) {
  v8h lo = *(const v8h*)p;
  v8h hi = *(const v8h*)(p + 16);
  return __builtin_shufflevector(lo, hi, 0,1,2,3,4,5,6,7,8,9,10,11,12,13,14,15);
}
__device__ __forceinline__ v16h ldfragB(const half_t* p) {
  v8h lo = *(const v8h*)p;
  v8h hi = *(const v8h*)(p + 16);
  return __builtin_shufflevector(lo, hi, 0,1,2,3,4,5,6,7,8,9,10,11,12,13,14,15);
}

// ---------------------------------------------------------------------------
// Generic GEMM:  C[M,N] = A[M,K] (f16, row-major) @ Wt[N,K]^T (f16) + bias
// Tile 128x128, BK=64, 256 threads = 8 waves (2x4), each wave 64x32.
// Double-buffered LDS fed by async global->LDS copies.
// EPI: 0 = f16 out (* alpha), 1 = f16 out with exact GELU, 2 = f32 out.
// ---------------------------------------------------------------------------
template<int EPI>
__global__ __launch_bounds__(256)
void gemm_wmma(const half_t* __restrict__ A, const half_t* __restrict__ Bt,
               const float* __restrict__ bias, float* __restrict__ Cf,
               half_t* __restrict__ Ch, int M, int N, int K, float alpha)
{
  __shared__ __align__(32) half_t As[2][128*64];
  __shared__ __align__(32) half_t Bs[2][128*64];
  const int tid   = threadIdx.x;
  const int lane  = tid & 31, wid = tid >> 5;
  const int hlf   = lane >> 4, l16 = lane & 15;
  const int wr    = wid >> 2, wc = wid & 3;          // 2 x 4 wave grid
  const int blockM = blockIdx.y * 128;
  const int blockN = blockIdx.x * 128;

  v8f acc[4][2];
  #pragma unroll
  for (int i = 0; i < 4; ++i)
    #pragma unroll
    for (int j = 0; j < 2; ++j)
      #pragma unroll
      for (int e = 0; e < 8; ++e) acc[i][j][e] = 0.f;

  const int nt = K >> 6;                             // K / 64 tiles
  auto issue_tile = [&](int t, int buf) {
    const int k0 = t << 6;
    #pragma unroll
    for (int i = 0; i < 4; ++i) {
      int c = tid + 256*i;               // 1024 chunks of 8 halves per matrix
      int r = c >> 3, col = (c & 7) << 3;
      async_copy_b128(&As[buf][r*64 + col], A  + (size_t)(blockM + r)*K + k0 + col);
      async_copy_b128(&Bs[buf][r*64 + col], Bt + (size_t)(blockN + r)*K + k0 + col);
    }
  };

  issue_tile(0, 0);
  for (int t = 0; t < nt; ++t) {
    async_wait0();                 // this wave's copies for tile t are in LDS
    __syncthreads();               // all waves' copies done; prev compute done
    if (t + 1 < nt) issue_tile(t + 1, (t + 1) & 1);

    const half_t* Ab = As[t & 1];
    const half_t* Bb = Bs[t & 1];
    #pragma unroll
    for (int kk = 0; kk < 64; kk += 32) {
      v16h af[4], bf[2];
      #pragma unroll
      for (int fm = 0; fm < 4; ++fm)
        af[fm] = ldfragA(&Ab[(wr*64 + fm*16 + l16)*64 + kk + 8*hlf]);
      #pragma unroll
      for (int fn = 0; fn < 2; ++fn)
        bf[fn] = ldfragB(&Bb[(wc*32 + fn*16 + l16)*64 + kk + 8*hlf]);
      #pragma unroll
      for (int fm = 0; fm < 4; ++fm)
        #pragma unroll
        for (int fn = 0; fn < 2; ++fn)
          acc[fm][fn] = __builtin_amdgcn_wmma_f32_16x16x32_f16(
              false, af[fm], false, bf[fn], (short)0, acc[fm][fn], false, false);
    }
  }

  // Epilogue.  C/D layout: lane l -> col l%16, VGPR r -> row r + 8*(l/16).
  #pragma unroll
  for (int fm = 0; fm < 4; ++fm)
    #pragma unroll
    for (int fn = 0; fn < 2; ++fn)
      #pragma unroll
      for (int r = 0; r < 8; ++r) {
        int gm = blockM + wr*64 + fm*16 + r + 8*hlf;
        int gn = blockN + wc*32 + fn*16 + l16;
        float v = acc[fm][fn][r] + bias[gn];
        if constexpr (EPI == 2) {
          Cf[(size_t)gm*N + gn] = v;
        } else if constexpr (EPI == 1) {
          float g = 0.5f * v * (1.f + erff(v * 0.70710678118654752f));
          Ch[(size_t)gm*N + gn] = (half_t)g;
        } else {
          Ch[(size_t)gm*N + gn] = (half_t)(v * alpha);
        }
      }
}

// ---------------------------------------------------------------------------
// Flash attention: one block = one (b,h) and a 64-query tile; 4 waves,
// each wave owns 16 query rows.  Online softmax over 64-key tiles.
// ---------------------------------------------------------------------------
__global__ __launch_bounds__(128)
void flash_attn(const half_t* __restrict__ qh, const half_t* __restrict__ kh,
                const half_t* __restrict__ vh, const int* __restrict__ mask,
                half_t* __restrict__ ctxh)
{
  __shared__ __align__(32) half_t q_lds [64*64];   // [q][dh]
  __shared__ __align__(32) half_t k_lds [64*64];   // [kv][dh]
  __shared__ __align__(32) half_t vT_lds[64*64];   // [dh][kv]
  __shared__ __align__(32) half_t p_lds [64*64];   // [q][kv]
  const int tid  = threadIdx.x;
  const int lane = tid & 31, wid = tid >> 5;       // 4 waves
  const int hlf  = lane >> 4, l16 = lane & 15;
  const int b    = blockIdx.x / H_, h = blockIdx.x % H_;
  const int qblk = blockIdx.y * 64;

  #pragma unroll
  for (int i = 0; i < 4; ++i) {                    // stage 64x64 q tile (async)
    int c = tid + 128*i;
    int r = c >> 3, col = (c & 7) << 3;
    async_copy_b128(&q_lds[r*64 + col],
                    qh + ((size_t)(b*S_ + qblk + r))*D_ + h*DH_ + col);
  }

  float m_st[8], l_st[8];
  v8f   oacc[4];
  #pragma unroll
  for (int r = 0; r < 8; ++r) { m_st[r] = -1e30f; l_st[r] = 0.f; }
  #pragma unroll
  for (int fn = 0; fn < 4; ++fn)
    #pragma unroll
    for (int r = 0; r < 8; ++r) oacc[fn][r] = 0.f;

  for (int t0 = 0; t0 < S_; t0 += 64) {
    #pragma unroll
    for (int i = 0; i < 4; ++i) {                  // stage k tile (async) + vT
      int c = tid + 128*i;
      int r = c >> 3, col = (c & 7) << 3;
      size_t g = ((size_t)(b*S_ + t0 + r))*D_ + h*DH_ + col;
      async_copy_b128(&k_lds[r*64 + col], kh + g);
      v8h vv = *(const v8h*)(vh + g);
      #pragma unroll
      for (int j = 0; j < 8; ++j) vT_lds[(col + j)*64 + r] = vv[j];
    }
    async_wait0();
    __syncthreads();

    v8f sacc[4];                                   // scores: q @ k^T, K=DH
    #pragma unroll
    for (int fn = 0; fn < 4; ++fn)
      #pragma unroll
      for (int e = 0; e < 8; ++e) sacc[fn][e] = 0.f;
    #pragma unroll
    for (int kk = 0; kk < 64; kk += 32) {
      v16h aq = ldfragA(&q_lds[(wid*16 + l16)*64 + kk + 8*hlf]);
      #pragma unroll
      for (int fn = 0; fn < 4; ++fn) {
        v16h bk = ldfragB(&k_lds[(fn*16 + l16)*64 + kk + 8*hlf]);
        sacc[fn] = __builtin_amdgcn_wmma_f32_16x16x32_f16(
            false, aq, false, bk, (short)0, sacc[fn], false, false);
      }
    }

    // scale 1/sqrt(DH), mask, per-row max over 64 keys (16-lane halves)
    float mx[8];
    #pragma unroll
    for (int r = 0; r < 8; ++r) mx[r] = -1e30f;
    #pragma unroll
    for (int fn = 0; fn < 4; ++fn) {
      bool ok = mask[b*S_ + t0 + fn*16 + l16] != 0;
      #pragma unroll
      for (int r = 0; r < 8; ++r) {
        float sv = ok ? sacc[fn][r] * 0.125f : -1e30f;
        sacc[fn][r] = sv;
        mx[r] = fmaxf(mx[r], sv);
      }
    }
    #pragma unroll
    for (int r = 0; r < 8; ++r) {
      mx[r] = fmaxf(mx[r], __shfl_xor(mx[r], 8, 32));
      mx[r] = fmaxf(mx[r], __shfl_xor(mx[r], 4, 32));
      mx[r] = fmaxf(mx[r], __shfl_xor(mx[r], 2, 32));
      mx[r] = fmaxf(mx[r], __shfl_xor(mx[r], 1, 32));
    }
    float corr[8];
    #pragma unroll
    for (int r = 0; r < 8; ++r) {
      float mnew = fmaxf(m_st[r], mx[r]);
      corr[r] = expf(m_st[r] - mnew);
      m_st[r] = mnew;
      float rs = 0.f;
      #pragma unroll
      for (int fn = 0; fn < 4; ++fn) {
        float p = expf(sacc[fn][r] - mnew);
        sacc[fn][r] = p;
        rs += p;
      }
      rs += __shfl_xor(rs, 8, 32);
      rs += __shfl_xor(rs, 4, 32);
      rs += __shfl_xor(rs, 2, 32);
      rs += __shfl_xor(rs, 1, 32);
      l_st[r] = l_st[r] * corr[r] + rs;
    }
    #pragma unroll
    for (int fn = 0; fn < 4; ++fn)
      #pragma unroll
      for (int r = 0; r < 8; ++r) {
        p_lds[(wid*16 + r + 8*hlf)*64 + fn*16 + l16] = (half_t)sacc[fn][r];
        oacc[fn][r] *= corr[r];
      }
    __syncthreads();

    #pragma unroll
    for (int kk = 0; kk < 64; kk += 32) {          // ctx += P @ V, K = keys
      v16h ap = ldfragA(&p_lds[(wid*16 + l16)*64 + kk + 8*hlf]);
      #pragma unroll
      for (int fn = 0; fn < 4; ++fn) {
        v16h bv = ldfragB(&vT_lds[(fn*16 + l16)*64 + kk + 8*hlf]);
        oacc[fn] = __builtin_amdgcn_wmma_f32_16x16x32_f16(
            false, ap, false, bv, (short)0, oacc[fn], false, false);
      }
    }
    __syncthreads();
  }

  #pragma unroll
  for (int fn = 0; fn < 4; ++fn)
    #pragma unroll
    for (int r = 0; r < 8; ++r) {
      int qrow = qblk + wid*16 + r + 8*hlf;
      float v = oacc[fn][r] / l_st[r];
      ctxh[((size_t)(b*S_ + qrow))*D_ + h*DH_ + fn*16 + l16] = (half_t)v;
    }
}

// ---------------------------------------------------------------------------
// Row kernels: LayerNorms (block = one row of D=512, 256 threads x 2 elems)
// ---------------------------------------------------------------------------
__device__ __forceinline__ float blk_sum(float v, float* red) {
  const int t = threadIdx.x;
  red[t] = v;
  __syncthreads();
  #pragma unroll
  for (int s = 128; s > 0; s >>= 1) {
    if (t < s) red[t] += red[t + s];
    __syncthreads();
  }
  float r = red[0];
  __syncthreads();
  return r;
}

__global__ __launch_bounds__(256)
void embed_build_ln(const float* __restrict__ question, const float* __restrict__ vproj,
                    const float* __restrict__ pos_emb, const float* __restrict__ mod_emb,
                    const float* __restrict__ nv_g, const float* __restrict__ nv_b,
                    const float* __restrict__ emb_g, const float* __restrict__ emb_b,
                    float* __restrict__ x, half_t* __restrict__ xh)
{
  __shared__ float red[256];
  const int row = blockIdx.x;               // 0..BS-1
  const int b = row / S_, s = row % S_;
  const int d0 = threadIdx.x * 2;
  float y0, y1;
  if (s < Q_) {
    const float* qp = question + ((size_t)(b*Q_ + s))*D_;
    y0 = qp[d0]; y1 = qp[d0+1];
  } else {
    const float* vp = vproj + ((size_t)(b*T_ + (s - Q_)))*D_;
    y0 = vp[d0]; y1 = vp[d0+1];
    float mean = blk_sum(y0 + y1, red) * (1.f / D_);
    float a0 = y0 - mean, a1 = y1 - mean;
    float inv = rsqrtf(blk_sum(a0*a0 + a1*a1, red) * (1.f / D_) + EPS_);
    y0 = a0 * inv * nv_g[d0]   + nv_b[d0];
    y1 = a1 * inv * nv_g[d0+1] + nv_b[d0+1];
  }
  const int mo = (s < Q_) ? 0 : 1;
  y0 += pos_emb[(size_t)s*D_ + d0]   + mod_emb[(size_t)mo*D_ + d0];
  y1 += pos_emb[(size_t)s*D_ + d0+1] + mod_emb[(size_t)mo*D_ + d0+1];
  float mean = blk_sum(y0 + y1, red) * (1.f / D_);
  float a0 = y0 - mean, a1 = y1 - mean;
  float inv = rsqrtf(blk_sum(a0*a0 + a1*a1, red) * (1.f / D_) + EPS_);
  float o0 = a0 * inv * emb_g[d0]   + emb_b[d0];
  float o1 = a1 * inv * emb_g[d0+1] + emb_b[d0+1];
  size_t base = (size_t)row * D_;
  x[base + d0] = o0;  x[base + d0 + 1] = o1;
  xh[base + d0] = (half_t)o0;  xh[base + d0 + 1] = (half_t)o1;
}

__global__ __launch_bounds__(256)
void residual_ln(const float* __restrict__ gout, const float* __restrict__ g,
                 const float* __restrict__ bb, float* __restrict__ x,
                 half_t* __restrict__ xh)
{
  __shared__ float red[256];
  const int row = blockIdx.x;
  const int d0 = threadIdx.x * 2;
  size_t base = (size_t)row * D_;
  float y0 = gout[base + d0]     + x[base + d0];
  float y1 = gout[base + d0 + 1] + x[base + d0 + 1];
  float mean = blk_sum(y0 + y1, red) * (1.f / D_);
  float a0 = y0 - mean, a1 = y1 - mean;
  float inv = rsqrtf(blk_sum(a0*a0 + a1*a1, red) * (1.f / D_) + EPS_);
  float o0 = a0 * inv * g[d0]   + bb[d0];
  float o1 = a1 * inv * g[d0+1] + bb[d0+1];
  x[base + d0] = o0;  x[base + d0 + 1] = o1;
  xh[base + d0] = (half_t)o0;  xh[base + d0 + 1] = (half_t)o1;
}

// ---------------------------------------------------------------------------
// Conversions: f32 weights [K][N] -> f16 transposed [N][K]; f32 -> f16 copy.
// ---------------------------------------------------------------------------
__global__ void wconv_t(const float* __restrict__ W, half_t* __restrict__ Wt,
                        int K, int N) {
  int idx = blockIdx.x * 256 + threadIdx.x;
  if (idx < K * N) {
    int k = idx / N, n = idx % N;
    Wt[(size_t)n * K + k] = (half_t)W[idx];
  }
}
__global__ void f32_to_f16(const float* __restrict__ src,
                           half_t* __restrict__ dst, int n) {
  int idx = blockIdx.x * 256 + threadIdx.x;
  if (idx < n) dst[idx] = (half_t)src[idx];
}

// ---------------------------------------------------------------------------
extern "C" void kernel_launch(void* const* d_in, const int* in_sizes, int n_in,
                              void* d_out, int out_size, void* d_ws, size_t ws_size,
                              hipStream_t stream)
{
  (void)in_sizes; (void)n_in; (void)out_size; (void)ws_size;
  const float* video    = (const float*)d_in[0];
  const float* question = (const float*)d_in[1];
  const int*   mask     = (const int*)  d_in[2];
  const float* pos_emb  = (const float*)d_in[3];
  const float* mod_emb  = (const float*)d_in[4];
  const float* Wv   = (const float*)d_in[5];
  const float* bv   = (const float*)d_in[6];
  const float* nv_g = (const float*)d_in[7];
  const float* nv_b = (const float*)d_in[8];
  const float* emb_g = (const float*)d_in[9];
  const float* emb_b = (const float*)d_in[10];
  const float* Wq  = (const float*)d_in[11];
  const float* bq  = (const float*)d_in[12];
  const float* Wk  = (const float*)d_in[13];
  const float* bk  = (const float*)d_in[14];
  const float* Wva = (const float*)d_in[15];
  const float* bva = (const float*)d_in[16];
  const float* Wo  = (const float*)d_in[17];
  const float* bo  = (const float*)d_in[18];
  const float* ln1_g = (const float*)d_in[19];
  const float* ln1_b = (const float*)d_in[20];
  const float* W1  = (const float*)d_in[21];
  const float* b1  = (const float*)d_in[22];
  const float* W2  = (const float*)d_in[23];
  const float* b2  = (const float*)d_in[24];
  const float* ln2_g = (const float*)d_in[25];
  const float* ln2_b = (const float*)d_in[26];
  float* x = (float*)d_out;                       // x lives in d_out

  char* ws = (char*)d_ws;
  size_t off = 0;
  auto take = [&](size_t bytes) -> void* {
    void* p = ws + off;
    off = (off + bytes + 255) & ~(size_t)255;
    return p;
  };
  half_t* videoh = (half_t*)take((size_t)BT_*FD_*sizeof(half_t));
  half_t* wtv    = (half_t*)take((size_t)FD_*D_*sizeof(half_t));
  half_t *wtq[NL_], *wtk[NL_], *wtva[NL_], *wto[NL_], *wt1[NL_], *wt2[NL_];
  for (int i = 0; i < NL_; ++i) {
    wtq[i]  = (half_t*)take((size_t)D_*D_*sizeof(half_t));
    wtk[i]  = (half_t*)take((size_t)D_*D_*sizeof(half_t));
    wtva[i] = (half_t*)take((size_t)D_*D_*sizeof(half_t));
    wto[i]  = (half_t*)take((size_t)D_*D_*sizeof(half_t));
    wt1[i]  = (half_t*)take((size_t)D_*FF_*sizeof(half_t));
    wt2[i]  = (half_t*)take((size_t)FF_*D_*sizeof(half_t));
  }
  half_t* xh   = (half_t*)take((size_t)BS_*D_*sizeof(half_t));
  half_t* qh   = (half_t*)take((size_t)BS_*D_*sizeof(half_t));
  half_t* kh   = (half_t*)take((size_t)BS_*D_*sizeof(half_t));
  half_t* vh   = (half_t*)take((size_t)BS_*D_*sizeof(half_t));
  half_t* ctxh = (half_t*)take((size_t)BS_*D_*sizeof(half_t));
  half_t* hbuf = (half_t*)take((size_t)BS_*FF_*sizeof(half_t));
  float*  tmp  = (float*) take((size_t)BS_*D_*sizeof(float));

  { int n = BT_*FD_;
    f32_to_f16<<<(n + 255)/256, 256, 0, stream>>>(video, videoh, n); }
  wconv_t<<<(FD_*D_ + 255)/256, 256, 0, stream>>>(Wv, wtv, FD_, D_);
  for (int i = 0; i < NL_; ++i) {
    wconv_t<<<(D_*D_  + 255)/256, 256, 0, stream>>>(Wq  + (size_t)i*D_*D_,  wtq[i],  D_, D_);
    wconv_t<<<(D_*D_  + 255)/256, 256, 0, stream>>>(Wk  + (size_t)i*D_*D_,  wtk[i],  D_, D_);
    wconv_t<<<(D_*D_  + 255)/256, 256, 0, stream>>>(Wva + (size_t)i*D_*D_,  wtva[i], D_, D_);
    wconv_t<<<(D_*D_  + 255)/256, 256, 0, stream>>>(Wo  + (size_t)i*D_*D_,  wto[i],  D_, D_);
    wconv_t<<<(D_*FF_ + 255)/256, 256, 0, stream>>>(W1  + (size_t)i*D_*FF_, wt1[i],  D_, FF_);
    wconv_t<<<(FF_*D_ + 255)/256, 256, 0, stream>>>(W2  + (size_t)i*FF_*D_, wt2[i],  FF_, D_);
  }

  // video projection -> tmp (f32), then fused embed+LN -> x, xh
  gemm_wmma<2><<<dim3(D_/128, BT_/128), 256, 0, stream>>>(
      videoh, wtv, bv, tmp, nullptr, BT_, D_, FD_, 1.f);
  embed_build_ln<<<BS_, 256, 0, stream>>>(
      question, tmp, pos_emb, mod_emb, nv_g, nv_b, emb_g, emb_b, x, xh);

  for (int i = 0; i < NL_; ++i) {
    gemm_wmma<0><<<dim3(D_/128, BS_/128), 256, 0, stream>>>(
        xh, wtq[i], bq + i*D_, nullptr, qh, BS_, D_, D_, 1.f);
    gemm_wmma<0><<<dim3(D_/128, BS_/128), 256, 0, stream>>>(
        xh, wtk[i], bk + i*D_, nullptr, kh, BS_, D_, D_, 1.f);
    gemm_wmma<0><<<dim3(D_/128, BS_/128), 256, 0, stream>>>(
        xh, wtva[i], bva + i*D_, nullptr, vh, BS_, D_, D_, 1.f);
    flash_attn<<<dim3(B_*H_, S_/64), 128, 0, stream>>>(qh, kh, vh, mask, ctxh);
    gemm_wmma<2><<<dim3(D_/128, BS_/128), 256, 0, stream>>>(
        ctxh, wto[i], bo + i*D_, tmp, nullptr, BS_, D_, D_, 1.f);
    residual_ln<<<BS_, 256, 0, stream>>>(tmp, ln1_g + i*D_, ln1_b + i*D_, x, xh);
    gemm_wmma<1><<<dim3(FF_/128, BS_/128), 256, 0, stream>>>(
        xh, wt1[i], b1 + i*FF_, nullptr, hbuf, BS_, FF_, D_, 1.f);
    gemm_wmma<2><<<dim3(D_/128, BS_/128), 256, 0, stream>>>(
        hbuf, wt2[i], b2 + i*D_, tmp, nullptr, BS_, D_, FF_, 1.f);
    residual_ln<<<BS_, 256, 0, stream>>>(tmp, ln2_g + i*D_, ln2_b + i*D_, x, xh);
  }
}
